// SelfAttention2D_56736517980250
// MI455X (gfx1250) — matrix operand order repaired and checked by reading, not compiled, not run on hardware
//
#include <hip/hip_runtime.h>

// ---------------- types ----------------
typedef __attribute__((ext_vector_type(16))) __bf16       bf16x16;
typedef __attribute__((ext_vector_type(8)))  float        f32x8;
typedef __attribute__((ext_vector_type(4)))  unsigned int u32x4;
typedef __attribute__((ext_vector_type(4)))  int          i32x4;

typedef __attribute__((address_space(1))) i32x4 glb_i32x4;
typedef __attribute__((address_space(3))) i32x4 lds_i32x4;

union FragU { u32x4 q[2]; bf16x16 v; };

// dims
#define BB 4
#define CC 256
#define NN 4096           // 64*64
#define SCALE 0.0625f     // C^-0.5
#define L2E 1.4426950408889634f

// ---- CDNA5 async global->LDS path (guarded; falls back to reg bounce) ----
#if defined(__AMDGCN__) && __has_builtin(__builtin_amdgcn_global_load_async_to_lds_b128) && __has_builtin(__builtin_amdgcn_s_wait_asynccnt)
#define USE_ASYNC 1
#define WAIT_ASYNC(n) __builtin_amdgcn_s_wait_asynccnt(n)
#else
#define USE_ASYNC 0
#define WAIT_ASYNC(n)
#endif

static __device__ __forceinline__ void async_cp16(const unsigned short* g, unsigned short* l) {
#if USE_ASYNC
    __builtin_amdgcn_global_load_async_to_lds_b128(
        (glb_i32x4*)(i32x4*)(unsigned short*)g, (lds_i32x4*)(i32x4*)l, 0, 0);
#else
    *(u32x4*)l = *(const u32x4*)g;
#endif
}

static __device__ __forceinline__ unsigned short f2bf(float f) {
    unsigned int u = __float_as_uint(f);
    u += 0x7FFFu + ((u >> 16) & 1u);          // round-to-nearest-even
    return (unsigned short)(u >> 16);
}

// A-fragment (16x32 bf16): per lane, 16 contiguous halves along K
static __device__ __forceinline__ bf16x16 ld_frag_a(const unsigned short* p) {
    FragU u;
    u.q[0] = *(const u32x4*)(p);
    u.q[1] = *(const u32x4*)(p + 8);
    return u.v;
}
// B-fragment (32x16 bf16): per lane, 16 contiguous halves along K
static __device__ __forceinline__ bf16x16 ld_frag_b(const unsigned short* p) {
    FragU u;
    u.q[0] = *(const u32x4*)(p);
    u.q[1] = *(const u32x4*)(p + 8);
    return u.v;
}

static __device__ __forceinline__ f32x8 wmma_bf16(bf16x16 a, bf16x16 b, f32x8 c) {
    return __builtin_amdgcn_wmma_f32_16x16x32_bf16(false, a, false, b, (short)0, c, false, false);
}

// ---------------- kernel 1: x + positional encoding -> X (B,N,C) bf16 ----------------
__global__ __launch_bounds__(256) void k_prep_x(const float* __restrict__ x,
                                                unsigned short* __restrict__ X) {
    int idx = blockIdx.x * 256 + threadIdx.x;     // B*N*C threads, c fastest
    int c = idx & 255;
    int n = (idx >> 8) & (NN - 1);
    int b = idx >> 20;
    int h = n >> 6, w = n & 63;
    int cm = c & 63;
    float inv = exp2f(-(float)cm * 0.20762050593046f);   // 10000^(-cm/64)
    float t   = (c < 128) ? (float)w : (float)h;
    float ang = t * inv;
    float pe  = (c & 64) ? cosf(ang) : sinf(ang);
    float v   = x[(((b << 8) | c) << 12) + n] + pe;       // x[b][c][n]
    X[idx] = f2bf(v);                                     // X[b][n][c]
}

// ---------------- kernel 2: weights fp32 -> bf16 ----------------
__global__ __launch_bounds__(256) void k_prep_w(const float* __restrict__ wq,
                                                const float* __restrict__ wk,
                                                const float* __restrict__ wv,
                                                unsigned short* __restrict__ Wb) {
    int idx = blockIdx.x * 256 + threadIdx.x;     // 3*65536
    int sel = idx >> 16;
    int i   = idx & 65535;
    float v = (sel == 0) ? wq[i] : (sel == 1) ? wk[i] : wv[i];
    Wb[idx] = f2bf(v);
}

// ---------------- kernel 3: QKV projection (WMMA) ----------------
// Q,K row-major (B,N,C) bf16 ; V stored transposed Vt (B,C,N) bf16
__global__ __launch_bounds__(128) void k_qkv(const unsigned short* __restrict__ X,
                                             const unsigned short* __restrict__ Wb,
                                             unsigned short* __restrict__ Q,
                                             unsigned short* __restrict__ K,
                                             unsigned short* __restrict__ Vt) {
    int b    = blockIdx.x >> 6;
    int rb   = blockIdx.x & 63;
    int wave = threadIdx.x >> 5;
    int lane = threadIdx.x & 31;
    int lh   = lane & 15;
    int hi   = (lane >= 16) ? 1 : 0;
    int m0   = rb * 64 + wave * 16;

    const unsigned short* Xb   = X + ((size_t)b << 20);
    const unsigned short* Arow = Xb + (m0 + lh) * CC + (hi ? 16 : 0);

    for (int nt = 0; nt < 16; ++nt) {
        f32x8 acc0 = 0.0f, acc1 = 0.0f, acc2 = 0.0f;
        const unsigned short* Brow = Wb + (nt * 16 + lh) * CC + (hi ? 16 : 0);
#pragma unroll
        for (int kc = 0; kc < 8; ++kc) {
            bf16x16 a = ld_frag_a(Arow + kc * 32);
            acc0 = wmma_bf16(a, ld_frag_b(Brow + kc * 32),              acc0);
            acc1 = wmma_bf16(a, ld_frag_b(Brow + 65536 + kc * 32),      acc1);
            acc2 = wmma_bf16(a, ld_frag_b(Brow + 2 * 65536 + kc * 32),  acc2);
        }
        unsigned short* Qb = Q + ((size_t)b << 20);
        unsigned short* Kb = K + ((size_t)b << 20);
        unsigned short* Vb = Vt + ((size_t)b << 20);
        int col = nt * 16 + lh;
#pragma unroll
        for (int r = 0; r < 8; ++r) {
            int row = m0 + r + (hi ? 8 : 0);
            Qb[row * CC + col] = f2bf(acc0[r]);
            Kb[row * CC + col] = f2bf(acc1[r]);
            Vb[col * NN + row] = f2bf(acc2[r]);      // transposed store
        }
    }
}

// ---------------- kernel 4: flash attention with async LDS staging ----------------
__global__ __launch_bounds__(128) void k_attn(const unsigned short* __restrict__ Q,
                                              const unsigned short* __restrict__ Kk,
                                              const unsigned short* __restrict__ Vt,
                                              float* __restrict__ out) {
    // double-buffered K (64x256) and V (256x64) tiles + P transpose scratch
    __shared__ __align__(16) unsigned short KtS[2][64 * CC];
    __shared__ __align__(16) unsigned short VtS[2][CC * 64];
    __shared__ __align__(16) unsigned short Plds[4][16 * 64];

    int b    = blockIdx.x >> 6;
    int rb   = blockIdx.x & 63;
    int wave = threadIdx.x >> 5;
    int lane = threadIdx.x & 31;
    int tid  = threadIdx.x;
    int lh   = lane & 15;
    int hi   = (lane >= 16) ? 1 : 0;
    int m0   = rb * 64 + wave * 16;

    const unsigned short* Qrow = Q  + ((size_t)b << 20) + (m0 + lh) * CC + (hi ? 16 : 0);
    const unsigned short* Kg   = Kk + ((size_t)b << 20);
    const unsigned short* Vg   = Vt + ((size_t)b << 20);
    unsigned short*       Pw   = Plds[wave];

    // stage K tile (contiguous 32KB) + V tile (256 rows x 128B) for block kb
    auto stage_kv = [&](int kb, int buf) {
        const unsigned short* ks = Kg + kb * (64 * CC);
        unsigned short* kd = KtS[buf];
        unsigned short* vd = VtS[buf];
#pragma unroll
        for (int i = 0; i < 16; ++i) {                   // 16 x 2KB => 32KB
            int off = (tid + i * 128) * 8;               // halves, 16B chunks
            async_cp16(ks + off, kd + off);
        }
#pragma unroll
        for (int i = 0; i < 16; ++i) {                   // 2048 chunks of 16B
            int j = tid + i * 128;
            int d = j >> 3, sub = j & 7;                 // 8 chunks per Vt row
            async_cp16(Vg + d * NN + kb * 64 + sub * 8, vd + j * 8);
        }
    };

    f32x8 O[16];
#pragma unroll
    for (int dt = 0; dt < 16; ++dt) O[dt] = 0.0f;
    float mrow[8], lsum[8];
#pragma unroll
    for (int r = 0; r < 8; ++r) { mrow[r] = -INFINITY; lsum[r] = 0.0f; }

    stage_kv(0, 0);                                      // prime the pipeline

    for (int kb = 0; kb < 64; ++kb) {
        int cur = kb & 1;
        if (kb + 1 < 64) {                               // prefetch next block
            stage_kv(kb + 1, cur ^ 1);
            WAIT_ASYNC(32);                              // oldest 32 (cur) done
        } else {
            WAIT_ASYNC(0);
        }
        __syncthreads();                                 // tiles visible to all waves

        const unsigned short* Kl = KtS[cur];
        const unsigned short* Vl = VtS[cur];

        // ---- S = Q K^T over this 64-key block (4 tiles of 16x16) ----
        f32x8 S[4];
#pragma unroll
        for (int nt = 0; nt < 4; ++nt) S[nt] = 0.0f;
#pragma unroll
        for (int kc = 0; kc < 8; ++kc) {
            bf16x16 a = ld_frag_a(Qrow + kc * 32);
#pragma unroll
            for (int nt = 0; nt < 4; ++nt) {
                const unsigned short* kp = Kl + (nt * 16 + lh) * CC + kc * 32 + (hi ? 16 : 0);
                S[nt] = wmma_bf16(a, ld_frag_b(kp), S[nt]);
            }
        }

        // ---- online softmax (rows live across 16-lane halves) ----
        float alpha[8];
#pragma unroll
        for (int r = 0; r < 8; ++r) {
            float c0 = fmaxf(fmaxf(S[0][r], S[1][r]), fmaxf(S[2][r], S[3][r])) * SCALE;
            c0 = fmaxf(c0, __shfl_xor(c0, 1, 32));
            c0 = fmaxf(c0, __shfl_xor(c0, 2, 32));
            c0 = fmaxf(c0, __shfl_xor(c0, 4, 32));
            c0 = fmaxf(c0, __shfl_xor(c0, 8, 32));
            float mn = fmaxf(mrow[r], c0);
            alpha[r] = exp2f((mrow[r] - mn) * L2E);
            mrow[r]  = mn;
            float rs = 0.0f;
            int rowl = r + (hi ? 8 : 0);
#pragma unroll
            for (int nt = 0; nt < 4; ++nt) {
                float p = exp2f((S[nt][r] * SCALE - mn) * L2E);
                rs += p;
                Pw[rowl * 64 + nt * 16 + lh] = f2bf(p);
            }
            rs += __shfl_xor(rs, 1, 32);
            rs += __shfl_xor(rs, 2, 32);
            rs += __shfl_xor(rs, 4, 32);
            rs += __shfl_xor(rs, 8, 32);
            lsum[r] = lsum[r] * alpha[r] + rs;
        }
#pragma unroll
        for (int dt = 0; dt < 16; ++dt)
#pragma unroll
            for (int r = 0; r < 8; ++r) O[dt][r] *= alpha[r];

        __syncthreads();   // P (D-layout -> A-layout transpose) visible

        // ---- O += P V : P A-frags from LDS, V B-frags from LDS tile ----
#pragma unroll
        for (int kc = 0; kc < 2; ++kc) {
            bf16x16 pa = ld_frag_a(Pw + lh * 64 + kc * 32 + (hi ? 16 : 0));
#pragma unroll
            for (int dt = 0; dt < 16; ++dt) {
                const unsigned short* vp = Vl + (dt * 16 + lh) * 64 + kc * 32 + (hi ? 16 : 0);
                O[dt] = wmma_bf16(pa, ld_frag_b(vp), O[dt]);
            }
        }
        __syncthreads();   // done with cur buffer + Pw before next iteration
    }

    // ---- epilogue: normalize and store out (B,C,H,W) fp32 ----
#pragma unroll
    for (int dt = 0; dt < 16; ++dt) {
        int d = dt * 16 + lh;
#pragma unroll
        for (int r = 0; r < 8; ++r) {
            int row = m0 + r + (hi ? 8 : 0);
            out[(((size_t)b * CC) + d) * NN + row] = O[dt][r] / lsum[r];
        }
    }
}

// ---------------- host launcher ----------------
extern "C" void kernel_launch(void* const* d_in, const int* in_sizes, int n_in,
                              void* d_out, int out_size, void* d_ws, size_t ws_size,
                              hipStream_t stream) {
    (void)in_sizes; (void)n_in; (void)out_size; (void)ws_size;
    const float* x  = (const float*)d_in[0];
    const float* wq = (const float*)d_in[1];
    const float* wk = (const float*)d_in[2];
    const float* wv = (const float*)d_in[3];
    float* out = (float*)d_out;

    char* ws = (char*)d_ws;
    const size_t MB8 = (size_t)8 << 20;
    unsigned short* X  = (unsigned short*)(ws);              // (B,N,C) bf16, 8MB
    unsigned short* Q  = (unsigned short*)(ws + MB8);        // 8MB
    unsigned short* K  = (unsigned short*)(ws + 2 * MB8);    // 8MB
    unsigned short* Vt = (unsigned short*)(ws + 3 * MB8);    // (B,C,N) bf16, 8MB
    unsigned short* Wb = (unsigned short*)(ws + 4 * MB8);    // 3*256*256 bf16

    k_prep_x<<<dim3((BB * NN * CC) / 256), dim3(256), 0, stream>>>(x, X);
    k_prep_w<<<dim3((3 * 65536) / 256), dim3(256), 0, stream>>>(wq, wk, wv, Wb);
    k_qkv <<<dim3(BB * (NN / 64)), dim3(128), 0, stream>>>(X, Wb, Q, K, Vt);
    k_attn<<<dim3(BB * (NN / 64)), dim3(128), 0, stream>>>(Q, K, Vt, out);
}
